// NGramContext_27470610825462
// MI455X (gfx1250) — matrix operand order repaired and checked by reading, not compiled, not run on hardware
//
#include <hip/hip_runtime.h>
#include <hip/hip_bf16.h>

typedef __attribute__((ext_vector_type(16))) __bf16 v16bf;
typedef __attribute__((ext_vector_type(8)))  float  v8f;

// ---------- bf16 helpers (round-to-nearest-even) ----------
static __device__ __forceinline__ unsigned short f2bf(float f) {
  union { float f; unsigned int u; } c; c.f = f;
  unsigned int u = c.u + 0x7FFFu + ((c.u >> 16) & 1u);
  return (unsigned short)(u >> 16);
}
static __device__ __forceinline__ float bf2f(unsigned short h) {
  union { unsigned int u; float f; } c; c.u = ((unsigned int)h) << 16;
  return c.f;
}

struct Frag32B { uint4 lo, hi; };
static __device__ __forceinline__ v16bf make_frag(uint4 lo, uint4 hi) {
  Frag32B p{lo, hi};
  return __builtin_bit_cast(v16bf, p);
}

// A-matrix (16x32 bf16, M x K) fragment per ISA layout:
// lane m = lane&15 ; klo = lane<16 ? 0 : 8 ; halves at [klo, klo+8) and [klo+16, klo+24)
static __device__ __forceinline__ v16bf load_a(const unsigned short* base, int stride,
                                               int m, int k0, int lane) {
  int klo = (lane < 16) ? 0 : 8;
  const unsigned short* p = base + m * stride + k0 + klo;
  uint4 lo = *(const uint4*)(p);
  uint4 hi = *(const uint4*)(p + 16);
  return make_frag(lo, hi);
}

// B-matrix (32x16 bf16, K x N) fragment, B[k][n] = W[n][k] (W row-major [N][K]):
// lane n = lane&15 ; klo = lane<16 ? 0 : 8 ; halves at [klo, klo+8) and [klo+16, klo+24)
static __device__ __forceinline__ v16bf load_b(const unsigned short* W, int ldk,
                                               int n, int k0, int lane) {
  int koff = (lane >= 16) ? 8 : 0;
  const unsigned short* p = W + n * ldk + k0 + koff;
  uint4 lo = *(const uint4*)(p);
  uint4 hi = *(const uint4*)(p + 16);
  return make_frag(lo, hi);
}

static __device__ __forceinline__ v8f wmma_bf16(v16bf a, v16bf b, v8f c) {
  return __builtin_amdgcn_wmma_f32_16x16x32_bf16(false, a, false, b, (short)0, c,
                                                 false, false);
}

// ---------- CDNA5 async global->LDS copy (ASYNCcnt path) ----------
// LDS byte address is taken from the VDST operand; generic shared ptr low 32 bits = LDS byte offset.
static __device__ __forceinline__ void async_copy_b128(void* lds_dst, const void* gsrc) {
  unsigned ldsoff = (unsigned)(size_t)lds_dst;
  asm volatile("global_load_async_to_lds_b128 %0, %1, off"
               :: "v"(ldsoff), "v"(gsrc) : "memory");
}
static __device__ __forceinline__ void wait_async_zero() {
  asm volatile("s_wait_asynccnt 0" ::: "memory");
}

// ---------- kernel 0a: f32 -> bf16 cast ----------
__global__ void cvt_bf16_kernel(const float* __restrict__ src,
                                unsigned short* __restrict__ dst, int n) {
  int i = blockIdx.x * blockDim.x + threadIdx.x;
  if (i < n) dst[i] = f2bf(src[i]);
}

// ---------- kernel 0b: depthwise weight transpose (C,2,8,8) -> (p=64,g=2,C) f32 ----------
__global__ void uniw_tr_kernel(const float* __restrict__ w, float* __restrict__ wT) {
  int tid = blockIdx.x * blockDim.x + threadIdx.x;   // tid = p*512 + g*256 + c
  if (tid >= 32768) return;
  int c = tid & 255;
  int g = (tid >> 8) & 1;
  int p = tid >> 9;
  wT[tid] = w[(c * 2 + g) * 64 + p];
}

// ---------- kernel 1: depthwise conv 8x8/stride8, groups of 2 -> bf16 uni ----------
__global__ __launch_bounds__(256) void conv_kernel(const float* __restrict__ x,
                                                   const float* __restrict__ wT,
                                                   const float* __restrict__ ub,
                                                   unsigned short* __restrict__ uni) {
  int pos = blockIdx.x;               // (b, oh, ow)
  int b = pos >> 10, oh = (pos >> 5) & 31, ow = pos & 31;
  int c = threadIdx.x;                // channel 0..255
  float acc = ub[c];
  const float* xb = x + ((size_t)(b * 256 + oh * 8) * 256 + ow * 8) * 512;
  #pragma unroll 4
  for (int p = 0; p < 64; ++p) {
    int i = p >> 3, j = p & 7;
    float2 xv = *(const float2*)(xb + (size_t)(i * 256 + j) * 512 + 2 * c);
    acc += xv.x * wT[(p * 2 + 0) * 256 + c] + xv.y * wT[(p * 2 + 1) * 256 + c];
  }
  uni[(size_t)pos * 256 + c] = f2bf(acc);
}

// ---------- kernel 2: gather + QKV WMMA + attention + proj WMMA + token mean ----------
#define T_STRIDE   264   // 256 + 8 halves pad
#define QKV_STRIDE 776   // 768 + 8 halves pad
__global__ __launch_bounds__(256) void ngram_attn_kernel(
    const unsigned short* __restrict__ uni,
    const unsigned short* __restrict__ qkvw,
    const float* __restrict__ qkvb,
    const float* __restrict__ relb,
    const unsigned short* __restrict__ projw,
    const float* __restrict__ projb,
    unsigned short* __restrict__ ctx) {
  extern __shared__ unsigned short smem2[];
  unsigned short* T   = smem2;                       // 64 x T_STRIDE (tokens, later O)
  unsigned short* QKV = smem2 + 64 * T_STRIDE;       // 64 x QKV_STRIDE

  const int tid  = threadIdx.x;
  const int lane = tid & 31;
  const int wave = tid >> 5;
  const int pos_base = blockIdx.x * 16;
  const int dir = blockIdx.y;
  const int b   = pos_base >> 10;
  const int oh  = (pos_base >> 5) & 31;
  const int ow0 = pos_base & 31;

  // Phase A: gather 16 windows x 4 tokens (reflect padding per pad_fwd/pad_bwd)
  // via async global->LDS copies (ASYNCcnt), no VGPR round-trip.
  {
    int row = tid >> 2, q4 = tid & 3;
    int pos_l = row >> 2, tok = row & 3;
    int i = tok >> 1, j = tok & 1;
    int ow = ow0 + pos_l;
    int r, s;
    if (dir == 0) { r = oh + i; if (r >= 32) r = 30; s = ow + j; if (s >= 32) s = 30; }
    else          { r = oh + i; r = (r == 0) ? 1 : r - 1;
                    s = ow + j; s = (s == 0) ? 1 : s - 1; }
    const unsigned short* src = uni + (((b * 32 + r) * 32 + s) << 8) + q4 * 64;
    unsigned short* dst = T + row * T_STRIDE + q4 * 64;
    #pragma unroll
    for (int u = 0; u < 8; ++u)
      async_copy_b128(dst + u * 8, src + u * 8);
    wait_async_zero();
  }
  __syncthreads();

  // Phase B: QKV = T(64x256) @ qkv_w^T(256x768), bf16 WMMA, f32 accum
  {
    int rt = wave & 3, cw = wave >> 2;
    int m0 = rt * 16;
    for (int it = 0; it < 24; ++it) {
      int n0 = (cw * 24 + it) * 16;
      v8f c = {};
      #pragma unroll
      for (int k0 = 0; k0 < 256; k0 += 32) {
        v16bf a  = load_a(T, T_STRIDE, m0 + (lane & 15), k0, lane);
        v16bf bf = load_b(qkvw, 256, n0 + (lane & 15), k0, lane);
        c = wmma_bf16(a, bf, c);
      }
      int n = n0 + (lane & 15);
      float bias = qkvb[n];
      int mbase = m0 + ((lane < 16) ? 0 : 8);
      #pragma unroll
      for (int r = 0; r < 8; ++r)
        QKV[(mbase + r) * QKV_STRIDE + n] = f2bf(c[r] + bias);
    }
  }
  __syncthreads();

  // Phase C: per-(window, head) 4x4 attention in VALU; O written into T
  {
    const float scale = 0.17677669529663687f;   // 32^-0.5
    int task = tid >> 1, half = tid & 1;
    int pos_l = task >> 3, h = task & 7;
    const unsigned short* base = QKV + (pos_l * 4) * QKV_STRIDE + h * 32;
    float attnp[2][4];
    #pragma unroll
    for (int hh = 0; hh < 2; ++hh) {
      int nr = half * 2 + hh;
      float s[4];
      #pragma unroll
      for (int m = 0; m < 4; ++m) {
        float acc = 0.f;
        const unsigned short* qp = base + nr * QKV_STRIDE;
        const unsigned short* kp = base + m * QKV_STRIDE + 256;
        #pragma unroll
        for (int d = 0; d < 32; ++d) acc += bf2f(qp[d]) * bf2f(kp[d]);
        int ni = nr >> 1, nj = nr & 1, mi = m >> 1, mj = m & 1;
        int ridx = (ni - mi + 1) * 3 + (nj - mj + 1);
        s[m] = acc * scale + relb[ridx * 8 + h];
      }
      float mx = fmaxf(fmaxf(s[0], s[1]), fmaxf(s[2], s[3]));
      float e0 = __expf(s[0] - mx), e1 = __expf(s[1] - mx);
      float e2 = __expf(s[2] - mx), e3 = __expf(s[3] - mx);
      float inv = 1.f / (e0 + e1 + e2 + e3);
      attnp[hh][0] = e0 * inv; attnp[hh][1] = e1 * inv;
      attnp[hh][2] = e2 * inv; attnp[hh][3] = e3 * inv;
    }
    #pragma unroll
    for (int hh = 0; hh < 2; ++hh) {
      int nr = half * 2 + hh;
      unsigned short* orow = T + (pos_l * 4 + nr) * T_STRIDE + h * 32;
      #pragma unroll
      for (int d = 0; d < 32; ++d) {
        float acc = 0.f;
        #pragma unroll
        for (int m = 0; m < 4; ++m)
          acc += attnp[hh][m] *
                 bf2f(QKV[(pos_l * 4 + m) * QKV_STRIDE + 512 + h * 32 + d]);
        orow[d] = f2bf(acc);
      }
    }
  }
  __syncthreads();

  // Phase D: P = O(64x256) @ proj_w^T(256x256) via WMMA; token-mean + proj_b -> ctx bf16
  {
    int rt = wave & 3, cw = wave >> 2;
    int m0 = rt * 16;
    for (int it = 0; it < 8; ++it) {
      int n0 = (cw * 8 + it) * 16;
      v8f c = {};
      #pragma unroll
      for (int k0 = 0; k0 < 256; k0 += 32) {
        v16bf a  = load_a(T, T_STRIDE, m0 + (lane & 15), k0, lane);
        v16bf bf = load_b(projw, 256, n0 + (lane & 15), k0, lane);
        c = wmma_bf16(a, bf, c);
      }
      int n = n0 + (lane & 15);
      float pb = projb[n];
      // rows r..r+3 of a 16-row tile are the 4 tokens of one window:
      float s0 = (c[0] + c[1] + c[2] + c[3]) * 0.25f + pb;  // window rt*4 + (lane<16?0:2)
      float s1 = (c[4] + c[5] + c[6] + c[7]) * 0.25f + pb;  // window +1
      int p0 = rt * 4 + ((lane < 16) ? 0 : 2);
      ctx[(size_t)(pos_base + p0)     * 512 + dir * 256 + n] = f2bf(s0);
      ctx[(size_t)(pos_base + p0 + 1) * 512 + dir * 256 + n] = f2bf(s1);
    }
  }
}

// ---------- kernel 3: out(8192x512) = ctx(8192x512) @ merge_w^T + merge_b ----------
#define B_STRIDE 520   // 512 + 8 halves pad
__global__ __launch_bounds__(256) void merge_kernel(
    const unsigned short* __restrict__ ctx,
    const unsigned short* __restrict__ mw,
    const float* __restrict__ mb,
    float* __restrict__ out) {
  extern __shared__ unsigned short smem3[];
  unsigned short* Bl = smem3;                 // 64 x B_STRIDE
  int tid = threadIdx.x, lane = tid & 31, wave = tid >> 5;
  int m_base = blockIdx.x * 128;
  int n_base = blockIdx.y * 64;
  // stage B panel (merge_w rows n_base..n_base+63) into LDS via async copies
  {
    int nl = tid >> 2, q4 = tid & 3;
    const unsigned short* src = mw + (size_t)(n_base + nl) * 512 + q4 * 128;
    unsigned short* dst = Bl + nl * B_STRIDE + q4 * 128;
    #pragma unroll
    for (int u = 0; u < 16; ++u)
      async_copy_b128(dst + u * 8, src + u * 8);
    wait_async_zero();
  }
  __syncthreads();
  int m0 = m_base + wave * 16;
  for (int ct = 0; ct < 4; ++ct) {
    int nl0 = ct * 16;
    v8f c = {};
    #pragma unroll 4
    for (int k0 = 0; k0 < 512; k0 += 32) {
      v16bf a  = load_a(ctx, 512, m0 + (lane & 15), k0, lane);
      v16bf bf = load_b(Bl, B_STRIDE, nl0 + (lane & 15), k0, lane);
      c = wmma_bf16(a, bf, c);
    }
    int n = n_base + nl0 + (lane & 15);
    float bias = mb[n];
    int mm = m0 + ((lane < 16) ? 0 : 8);
    #pragma unroll
    for (int r = 0; r < 8; ++r)
      out[(size_t)(mm + r) * 512 + n] = c[r] + bias;
  }
}

// ---------- host launch ----------
extern "C" void kernel_launch(void* const* d_in, const int* in_sizes, int n_in,
                              void* d_out, int out_size, void* d_ws, size_t ws_size,
                              hipStream_t stream) {
  const float* x        = (const float*)d_in[0];
  const float* uni_w    = (const float*)d_in[1];
  const float* uni_b    = (const float*)d_in[2];
  const float* qkv_w    = (const float*)d_in[3];
  const float* qkv_b    = (const float*)d_in[4];
  const float* rel_bias = (const float*)d_in[5];
  const float* proj_w   = (const float*)d_in[6];
  const float* proj_b   = (const float*)d_in[7];
  const float* merge_w  = (const float*)d_in[8];
  const float* merge_b  = (const float*)d_in[9];
  float* out = (float*)d_out;

  char* ws = (char*)d_ws;
  float*          wT       = (float*)(ws + 0);                 // 131072 B
  unsigned short* qkvw_bf  = (unsigned short*)(ws + 131072);   // 393216 B
  unsigned short* projw_bf = (unsigned short*)(ws + 524288);   // 131072 B
  unsigned short* mergew_bf= (unsigned short*)(ws + 655360);   // 524288 B
  unsigned short* uni_bf   = (unsigned short*)(ws + 1179648);  // 4194304 B
  unsigned short* ctx_bf   = (unsigned short*)(ws + 5373952);  // 8388608 B

  // weight casts + depthwise-weight transpose
  cvt_bf16_kernel<<<(196608 + 255) / 256, 256, 0, stream>>>(qkv_w, qkvw_bf, 196608);
  cvt_bf16_kernel<<<(65536 + 255) / 256, 256, 0, stream>>>(proj_w, projw_bf, 65536);
  cvt_bf16_kernel<<<(262144 + 255) / 256, 256, 0, stream>>>(merge_w, mergew_bf, 262144);
  uniw_tr_kernel<<<128, 256, 0, stream>>>(uni_w, wT);

  // depthwise conv (bandwidth-bound: 1.07 GB of x)
  conv_kernel<<<8192, 256, 0, stream>>>(x, wT, uni_b, uni_bf);

  // fused n-gram attention, fwd+bwd directions
  size_t smem2 = (size_t)(64 * T_STRIDE + 64 * QKV_STRIDE) * sizeof(unsigned short);
  ngram_attn_kernel<<<dim3(512, 2), 256, smem2, stream>>>(
      uni_bf, qkvw_bf, qkv_b, rel_bias, projw_bf, proj_b, ctx_bf);

  // merge GEMM
  size_t smem3 = (size_t)(64 * B_STRIDE) * sizeof(unsigned short);
  merge_kernel<<<dim3(64, 8), 256, smem3, stream>>>(ctx_bf, mergew_bf, merge_b, out);
}
